// DirectionalConv_83786222010845
// MI455X (gfx1250) — matrix operand order repaired and checked by reading, not compiled, not run on hardware
//
#include <hip/hip_runtime.h>

typedef __attribute__((ext_vector_type(16))) _Float16 v16h;
typedef __attribute__((ext_vector_type(8)))  _Float16 v8h;
typedef __attribute__((ext_vector_type(4)))  _Float16 v4h;
typedef __attribute__((ext_vector_type(8)))  float    v8f;

#define CIN    32
#define COUT   32
#define D_IN   48
#define H_IN   64
#define W_IN   128
#define D_OUT  48
#define H_OUT  68
#define W_OUT  128
#define NB     2
#define XP     132    // w extent with halo: w in [-2, 129] -> p in [0, 131]
#define XSTR   40     // padded f16 ci-stride in xlds
#define NTAPS  25
#define HCHUNK 4      // h rows per block (68 = 17 * 4)
#define W_ELTS (NTAPS * COUT * CIN)   // 25600
#define CI_STRIDE ((size_t)D_IN * H_IN * W_IN)   // x ci stride in floats

// ---- one-time prep: weight (co,ci,i,0,k) f32 -> w16[tap][co][ci] f16 ----
__global__ __launch_bounds__(256)
void prep_weights_kernel(const float* __restrict__ weight, _Float16* __restrict__ w16)
{
    int idx = blockIdx.x * 256 + threadIdx.x;   // = tap*1024 + co*32 + ci
    if (idx < W_ELTS) {
        int ci  = idx & 31;
        int co  = (idx >> 5) & 31;
        int tap = idx >> 10;
        w16[idx] = (_Float16)weight[(co * CIN + ci) * NTAPS + tap];
    }
}

// out[b,co,d,h,w] = sum_{ci,i,k} x[b,ci,d-2+i,h-2+i,w-2+k] * W[co,ci,i,0,k] + bias[co]
template <bool USE_WS>
__global__ __launch_bounds__(256)
void mdconv_wmma_kernel(const float* __restrict__ x,
                        const float* __restrict__ weight,
                        const _Float16* __restrict__ w16,
                        const float* __restrict__ bias,
                        float* __restrict__ out)
{
    __shared__ __align__(16) _Float16 wlds[W_ELTS];     // [tap][co][ci], 51200 B
    __shared__ __align__(16) _Float16 xlds[XP * XSTR];  // [p][ci] current row, 10560 B

    const int h0  = blockIdx.x * HCHUNK;  // 0,4,...,64
    const int d   = blockIdx.y;           // 0..47
    const int b   = blockIdx.z;           // 0..1
    const int tid = threadIdx.x;
    const int lane   = tid & 31;
    const int waveId = tid >> 5;          // 0..7 (wave32)

    // ---- stage f16 weights (once per block) ----
    if (USE_WS) {
        // coalesced b64 copy, 25 chunks in flight, conflict-free LDS writes
        const unsigned long long* src = (const unsigned long long*)w16;
        unsigned long long* dst = (unsigned long long*)wlds;
        #pragma unroll
        for (int j = 0; j < 25; ++j)
            dst[tid + 256 * j] = src[tid + 256 * j];
    } else {
        for (int idx = tid; idx < W_ELTS; idx += 256) {
            int ci  = idx & 31;
            int co  = (idx >> 5) & 31;
            int tap = idx >> 10;
            wlds[idx] = (_Float16)weight[(co * CIN + ci) * NTAPS + tap];
        }
    }

    // ---- zero the constant halo rows p in {0,1,130,131}, ci 0..31 (once) ----
    if (tid < 32) {
        int r = tid >> 3;                     // 0..3
        int c = tid & 7;                      // u64 chunk 0..7 (covers ci 0..31)
        int p = (r < 2) ? r : (128 + r);      // 0,1,130,131
        ((unsigned long long*)&xlds[p * XSTR])[c] = 0ull;
    }

    // per-wave tile assignment
    const int coBase = (waveId & 1) * 16;
    const int wBase  = (waveId >> 1) * 32;
    const int n      = lane & 15;
    const int hiHalf = lane >> 4;
    const int kA     = hiHalf * 16;   // A-frag ci base (contiguous k, matches B)
    const int kB     = hiHalf * 16;   // B-frag ci base (contiguous k, matches A)

    // bias per output row owned by this lane (C layout: M = v + 8*hiHalf)
    float bv[8];
    #pragma unroll
    for (int v = 0; v < 8; ++v) bv[v] = bias[coBase + v + 8 * hiHalf];

    for (int hc = 0; hc < HCHUNK; ++hc) {
        const int h = h0 + hc;
        v8f acc0 = {};
        v8f acc1 = {};

        for (int i = 0; i < 5; ++i) {
            __syncthreads();   // prior reads of xlds (and first-iter fills) complete
            const int dd = d - 2 + i;
            const int hh = h - 2 + i;
            const bool rowOK = (dd >= 0) && (dd < D_IN) && (hh >= 0) && (hh < H_IN);

            // stage interior: xlds[(w+2)*XSTR + ci] = f16(x[b,ci,dd,hh,w])
            // 1024 chunks of 4 ci; u = tid + 256*j; w = u&127 (coalesced), ci0 = (u>>7)*4
            if (rowOK) {
                const float* xrow = &x[((size_t)(b * CIN) * D_IN + dd) * (size_t)H_IN * W_IN
                                       + (size_t)hh * W_IN];
                float fbuf[16];
                #pragma unroll
                for (int j = 0; j < 4; ++j) {       // issue all 16 loads first
                    int u = tid + 256 * j;
                    int w = u & 127;
                    int ci0 = (u >> 7) * 4;
                    const float* p0 = xrow + (size_t)ci0 * CI_STRIDE + w;
                    #pragma unroll
                    for (int q = 0; q < 4; ++q)
                        fbuf[j * 4 + q] = p0[(size_t)q * CI_STRIDE];
                }
                #pragma unroll
                for (int j = 0; j < 4; ++j) {       // then convert + store
                    int u = tid + 256 * j;
                    int w = u & 127;
                    int ci0 = (u >> 7) * 4;
                    v4h hv;
                    #pragma unroll
                    for (int q = 0; q < 4; ++q) hv[q] = (_Float16)fbuf[j * 4 + q];
                    *(v4h*)&xlds[(w + 2) * XSTR + ci0] = hv;
                }
            } else {
                v4h z = {};
                #pragma unroll
                for (int j = 0; j < 4; ++j) {
                    int u = tid + 256 * j;
                    int w = u & 127;
                    int ci0 = (u >> 7) * 4;
                    *(v4h*)&xlds[(w + 2) * XSTR + ci0] = z;
                }
            }
            __syncthreads();

            #pragma unroll
            for (int k = 0; k < 5; ++k) {
                const int tap = i * 5 + k;
                // A fragment (weights): slots 0-15 -> ci kA..kA+15 contiguous
                const v8h* wp = reinterpret_cast<const v8h*>(
                    &wlds[tap * (COUT * CIN) + (coBase + n) * CIN + kA]);
                v8h alo = wp[0];
                v8h ahi = wp[1];   // +8 f16
                v16h afrag = __builtin_shufflevector(alo, ahi,
                    0,1,2,3,4,5,6,7,8,9,10,11,12,13,14,15);

                // B fragments (x): lane column w = wBase(+16)+n, ci = kB..kB+15 contiguous
                const v8h* xp0 = reinterpret_cast<const v8h*>(
                    &xlds[(wBase + n + k) * XSTR + kB]);
                v16h bfrag0 = __builtin_shufflevector(xp0[0], xp0[1],
                    0,1,2,3,4,5,6,7,8,9,10,11,12,13,14,15);
                const v8h* xp1 = reinterpret_cast<const v8h*>(
                    &xlds[(wBase + 16 + n + k) * XSTR + kB]);
                v16h bfrag1 = __builtin_shufflevector(xp1[0], xp1[1],
                    0,1,2,3,4,5,6,7,8,9,10,11,12,13,14,15);

                acc0 = __builtin_amdgcn_wmma_f32_16x16x32_f16(
                    false, afrag, false, bfrag0, (short)0, acc0, false, false);
                acc1 = __builtin_amdgcn_wmma_f32_16x16x32_f16(
                    false, afrag, false, bfrag1, (short)0, acc1, false, false);
            }
        }

        // ---- epilogue for this h: C/D layout VGPR v -> M = v + 8*hiHalf, N = lane&15
        const int wc0 = wBase + n;
        const int wc1 = wc0 + 16;
        #pragma unroll
        for (int v = 0; v < 8; ++v) {
            int co = coBase + v + 8 * hiHalf;
            size_t o = (((size_t)(b * COUT + co) * D_OUT + d) * H_OUT + h) * (size_t)W_OUT;
            out[o + wc0] = acc0[v] + bv[v];
            out[o + wc1] = acc1[v] + bv[v];
        }
    }
}

extern "C" void kernel_launch(void* const* d_in, const int* in_sizes, int n_in,
                              void* d_out, int out_size, void* d_ws, size_t ws_size,
                              hipStream_t stream) {
    const float* x      = (const float*)d_in[0];
    const float* weight = (const float*)d_in[1];
    const float* bias   = (const float*)d_in[2];
    float* out = (float*)d_out;

    dim3 grid(H_OUT / HCHUNK, D_OUT, NB);   // 17 x 48 x 2 = 1632 blocks

    const size_t wbytes = (size_t)W_ELTS * 2;   // 51200 B of f16 weights
    if (ws_size >= wbytes) {
        _Float16* w16 = (_Float16*)d_ws;
        prep_weights_kernel<<<(W_ELTS + 255) / 256, 256, 0, stream>>>(weight, w16);
        mdconv_wmma_kernel<true><<<grid, 256, 0, stream>>>(x, weight, w16, bias, out);
    } else {
        mdconv_wmma_kernel<false><<<grid, 256, 0, stream>>>(x, weight, nullptr, bias, out);
    }

    (void)in_sizes; (void)n_in; (void)out_size;
}
